// MoeMLP_21483426414709
// MI455X (gfx1250) — compile-verified, matrix operand order checked by reading, not run on hardware
//
#include <hip/hip_runtime.h>
#include <hip/hip_bf16.h>

// ---------------- problem constants ----------------
#define NTOK   2048
#define DMODEL 1024
#define NEXP   8
#define DFFN   2048
#define W1ROW  (NEXP * DFFN)   // 16384, row stride of w1 [D, E*DFFN]

// ---------------- tiling ----------------
#define MT   128               // token-rows per workgroup tile
#define NT   128               // output cols per workgroup tile
#define KT   32                // K step (matches 16x16x32 WMMA)
#define LDSS 40                // LDS row stride in bf16 elems (80B, 16B-aligned, bank-spread)

typedef __attribute__((ext_vector_type(16))) __bf16 v16bf;
typedef __attribute__((ext_vector_type(8)))  float  v8f;

// fp32 -> bf16, round-to-nearest-even
__device__ __forceinline__ __bf16 f2bf(float f) {
  union { float f; unsigned u; } c; c.f = f;
  unsigned r = c.u + 0x7FFFu + ((c.u >> 16) & 1u);
  unsigned short h = (unsigned short)(r >> 16);
  return __builtin_bit_cast(__bf16, h);
}

// Load one 16x32 bf16 WMMA operand fragment from an LDS tile stored so that
// each matrix row/col's 32 K-values are contiguous at p.
// Per ISA 7.12.2: lanes 0-15 hold K {0..7, 16..23}; lanes 16-31 hold K {8..15, 24..31}.
__device__ __forceinline__ v16bf frag_ld(const __bf16* p, int lane) {
  const int k0 = (lane & 16) ? 8 : 0;
  union { v16bf v; uint4 q[2]; } u;
  u.q[0] = *(const uint4*)(p + k0);        // 8 bf16 = 16B
  u.q[1] = *(const uint4*)(p + k0 + 16);
  return u.v;
}

__device__ __forceinline__ v8f wmma_bf16(v16bf a, v16bf b, v8f c) {
  return __builtin_amdgcn_wmma_f32_16x16x32_bf16(
      /*neg_a=*/false, a, /*neg_b=*/false, b,
      /*c_mod=*/(short)0, c, /*reuse_a=*/false, /*reuse_b=*/false);
}

// CDNA5 async global->LDS copy, 16B per lane, tracked by ASYNCcnt.
// lds_off = low 32 bits of generic LDS address.
__device__ __forceinline__ void async_copy_b128(void* lds_ptr, const void* gptr) {
  unsigned loff = (unsigned)(unsigned long long)lds_ptr;
  asm volatile("global_load_async_to_lds_b128 %0, %1, off"
               :: "v"(loff), "v"(gptr) : "memory");
}
__device__ __forceinline__ void wait_async0() {
  asm volatile("s_wait_asynccnt 0" ::: "memory");
}

// =====================================================================
// Phase 1: routing.  logits = x @ w_router^T, top-2 of softmax, renorm.
// Builds per-expert ascending lists of (slot = 2*token + k, gate).
// Single workgroup => fully deterministic list order.
// =====================================================================
__global__ __launch_bounds__(256) void moe_router(
    const float* __restrict__ x, const float* __restrict__ wr,
    int* __restrict__ sel_slot, float* __restrict__ sel_gate,
    int* __restrict__ cnt) {
  __shared__ float swr[NEXP * DMODEL];   // 32 KB
  __shared__ int   spk[NTOK];            // e0 | e1<<8
  __shared__ float sg0[NTOK];
  __shared__ float sg1[NTOK];

  const int tid = threadIdx.x;
  for (int i = tid; i < NEXP * DMODEL; i += 256) swr[i] = wr[i];
  __syncthreads();

  for (int t = tid; t < NTOK; t += 256) {
    float acc[NEXP];
#pragma unroll
    for (int e = 0; e < NEXP; ++e) acc[e] = 0.f;
    const float* xp = x + (size_t)t * DMODEL;
    for (int d = 0; d < DMODEL; d += 4) {
      float4 xv = *(const float4*)(xp + d);
#pragma unroll
      for (int e = 0; e < NEXP; ++e) {
        const float* w = swr + e * DMODEL + d;
        acc[e] += xv.x * w[0] + xv.y * w[1] + xv.z * w[2] + xv.w * w[3];
      }
    }
    // top-2 (first occurrence on ties, matching top_k)
    int i0 = 0; float m0 = acc[0];
#pragma unroll
    for (int e = 1; e < NEXP; ++e) if (acc[e] > m0) { m0 = acc[e]; i0 = e; }
    int i1 = (i0 == 0) ? 1 : 0; float m1 = acc[i1];
#pragma unroll
    for (int e = 0; e < NEXP; ++e)
      if (e != i0 && e != ((i0 == 0) ? 1 : 0) && acc[e] > m1) { m1 = acc[e]; i1 = e; }
    // renormalized softmax over the two kept logits
    float texp = expf(m1 - m0);          // <= 1
    float g0 = 1.f / (1.f + texp);
    float g1 = texp / (1.f + texp);
    spk[t] = i0 | (i1 << 8);
    sg0[t] = g0;
    sg1[t] = g1;
  }
  __syncthreads();

  if (tid < NEXP) {
    const int e = tid;
    int c = 0;
    for (int t = 0; t < NTOK; ++t) {
      int pk = spk[t];
      if ((pk & 0xFF) == e) {
        sel_slot[e * NTOK + c] = 2 * t;     sel_gate[e * NTOK + c] = sg0[t]; ++c;
      } else if ((pk >> 8) == e) {
        sel_slot[e * NTOK + c] = 2 * t + 1; sel_gate[e * NTOK + c] = sg1[t]; ++c;
      }
    }
    cnt[e] = c;
  }
}

// =====================================================================
// Phase 2: grouped GEMM1 + exact GELU.  H[slot, :] = gelu(x[tok] @ W1_e)
// grid = (Mtiles=16, experts=8, DFFN/NT=16)
// =====================================================================
__global__ __launch_bounds__(256) void moe_gemm1(
    const float* __restrict__ x, const float* __restrict__ w1,
    const int* __restrict__ sel_slot, const int* __restrict__ cnt,
    __bf16* __restrict__ hbuf) {
  const int e     = blockIdx.y;
  const int mbase = blockIdx.x * MT;
  const int nbase = blockIdx.z * NT;
  const int count = cnt[e];
  if (mbase >= count) return;

  __shared__ __bf16 lA[MT * LDSS];   // [row][k]   10 KB
  __shared__ __bf16 lB[NT * LDSS];   // [col][k]   10 KB (transposed on store)
  __shared__ int    srow[MT];        // slot id or -1

  const int tid  = threadIdx.x;
  const int lane = tid & 31;
  const int w    = tid >> 5;     // 8 waves
  const int wm   = w & 3;        // row quadrant: wm*32
  const int wn   = w >> 2;       // col half:     wn*64

  for (int r = tid; r < MT; r += 256) {
    int gi = mbase + r;
    srow[r] = (gi < count) ? sel_slot[e * NTOK + gi] : -1;
  }
  __syncthreads();

  v8f acc[2][4];
#pragma unroll
  for (int tm = 0; tm < 2; ++tm)
#pragma unroll
    for (int tn = 0; tn < 4; ++tn) acc[tm][tn] = (v8f){0.f,0.f,0.f,0.f,0.f,0.f,0.f,0.f};

  for (int kk = 0; kk < DMODEL; kk += KT) {
    __syncthreads();
    // stage A: gather token rows, fp32 -> bf16
    for (int idx = tid; idx < MT * (KT / 4); idx += 256) {   // 1024
      int r = idx >> 3, c4 = (idx & 7) << 2;
      int s = srow[r];
      float4 v = make_float4(0.f, 0.f, 0.f, 0.f);
      if (s >= 0) {
        const float* gp = x + (size_t)(s >> 1) * DMODEL + kk + c4;
        v = *(const float4*)gp;
      }
      __bf16* dst = lA + r * LDSS + c4;
      dst[0] = f2bf(v.x); dst[1] = f2bf(v.y); dst[2] = f2bf(v.z); dst[3] = f2bf(v.w);
    }
    // stage B transposed: lB[n][k], fp32 -> bf16; prefetch next K tile
    for (int idx = tid; idx < KT * (NT / 4); idx += 256) {   // 1024
      int k = idx >> 5, n4 = (idx & 31) << 2;
      const float* gp = w1 + (size_t)(kk + k) * W1ROW + e * DFFN + nbase + n4;
      float4 v = *(const float4*)gp;
      if (kk + KT < DMODEL) __builtin_prefetch(gp + (size_t)KT * W1ROW, 0, 1);
      lB[(n4 + 0) * LDSS + k] = f2bf(v.x);
      lB[(n4 + 1) * LDSS + k] = f2bf(v.y);
      lB[(n4 + 2) * LDSS + k] = f2bf(v.z);
      lB[(n4 + 3) * LDSS + k] = f2bf(v.w);
    }
    __syncthreads();

    v16bf a[2], b[4];
#pragma unroll
    for (int tm = 0; tm < 2; ++tm)
      a[tm] = frag_ld(lA + (wm * 32 + tm * 16 + (lane & 15)) * LDSS, lane);
#pragma unroll
    for (int tn = 0; tn < 4; ++tn)
      b[tn] = frag_ld(lB + (wn * 64 + tn * 16 + (lane & 15)) * LDSS, lane);
#pragma unroll
    for (int tm = 0; tm < 2; ++tm)
#pragma unroll
      for (int tn = 0; tn < 4; ++tn)
        acc[tm][tn] = wmma_bf16(a[tm], b[tn], acc[tm][tn]);
  }

  // epilogue: exact GELU, store bf16 H.  C layout: VGPR j -> row j (lanes 0-15)
  // or j+8 (lanes 16-31), col = lane%16.
  const int rhi = (lane & 16) ? 8 : 0;
#pragma unroll
  for (int tm = 0; tm < 2; ++tm) {
#pragma unroll
    for (int tn = 0; tn < 4; ++tn) {
      int col = nbase + wn * 64 + tn * 16 + (lane & 15);
#pragma unroll
      for (int j = 0; j < 8; ++j) {
        int rloc = wm * 32 + tm * 16 + j + rhi;
        int s = srow[rloc];
        if (s >= 0) {
          float h = acc[tm][tn][j];
          h = 0.5f * h * (1.f + erff(h * 0.70710678118654752f));
          hbuf[(size_t)s * DFFN + col] = f2bf(h);
        }
      }
    }
  }
}

// =====================================================================
// Phase 3: grouped GEMM2 + gate scale.  Y[slot, :] = gate * (H[slot] @ W2_e)
// A-operand rows are already bf16 in hbuf -> staged via CDNA5 async
// global->LDS DMA (ASYNCcnt), no VGPR round-trip.
// grid = (Mtiles=16, experts=8, DMODEL/NT=8)
// =====================================================================
__global__ __launch_bounds__(256) void moe_gemm2(
    const __bf16* __restrict__ hbuf, const float* __restrict__ w2,
    const int* __restrict__ sel_slot, const float* __restrict__ sel_gate,
    const int* __restrict__ cnt, float* __restrict__ ybuf) {
  const int e     = blockIdx.y;
  const int mbase = blockIdx.x * MT;
  const int nbase = blockIdx.z * NT;
  const int count = cnt[e];
  if (mbase >= count) return;

  __shared__ __bf16 lA[MT * LDSS];
  __shared__ __bf16 lB[NT * LDSS];
  __shared__ int    srow[MT];
  __shared__ float  sgate[MT];

  const int tid  = threadIdx.x;
  const int lane = tid & 31;
  const int w    = tid >> 5;
  const int wm   = w & 3;
  const int wn   = w >> 2;

  for (int r = tid; r < MT; r += 256) {
    int gi = mbase + r;
    if (gi < count) {
      srow[r]  = sel_slot[e * NTOK + gi];
      sgate[r] = sel_gate[e * NTOK + gi];
    } else {
      srow[r] = -1; sgate[r] = 0.f;
    }
  }
  __syncthreads();

  // zero-fill padding rows of lA once; the async path never writes them.
  for (int idx = tid; idx < MT * (KT / 8); idx += 256) {     // 512
    int r = idx >> 2, c8 = (idx & 3) << 3;
    if (srow[r] < 0) *(uint4*)(lA + r * LDSS + c8) = make_uint4(0u, 0u, 0u, 0u);
  }

  v8f acc[2][4];
#pragma unroll
  for (int tm = 0; tm < 2; ++tm)
#pragma unroll
    for (int tn = 0; tn < 4; ++tn) acc[tm][tn] = (v8f){0.f,0.f,0.f,0.f,0.f,0.f,0.f,0.f};

  for (int kk = 0; kk < DFFN; kk += KT) {
    __syncthreads();
    // stage A: gathered bf16 rows via async DMA into LDS (16B per lane)
    for (int idx = tid; idx < MT * (KT / 8); idx += 256) {   // 512
      int r = idx >> 2, c8 = (idx & 3) << 3;
      int s = srow[r];
      if (s >= 0) {
        async_copy_b128(lA + r * LDSS + c8,
                        hbuf + (size_t)s * DFFN + kk + c8);
      }
    }
    // stage B transposed: lB[n][k], fp32 -> bf16
    for (int idx = tid; idx < KT * (NT / 4); idx += 256) {   // 1024
      int k = idx >> 5, n4 = (idx & 31) << 2;
      const float* gp = w2 + (size_t)(e * DFFN + kk + k) * DMODEL + nbase + n4;
      float4 v = *(const float4*)gp;
      if (kk + KT < DFFN) __builtin_prefetch(gp + (size_t)KT * DMODEL, 0, 1);
      lB[(n4 + 0) * LDSS + k] = f2bf(v.x);
      lB[(n4 + 1) * LDSS + k] = f2bf(v.y);
      lB[(n4 + 2) * LDSS + k] = f2bf(v.z);
      lB[(n4 + 3) * LDSS + k] = f2bf(v.w);
    }
    wait_async0();       // this wave's async LDS writes are complete
    __syncthreads();     // all waves' writes complete

    v16bf a[2], b[4];
#pragma unroll
    for (int tm = 0; tm < 2; ++tm)
      a[tm] = frag_ld(lA + (wm * 32 + tm * 16 + (lane & 15)) * LDSS, lane);
#pragma unroll
    for (int tn = 0; tn < 4; ++tn)
      b[tn] = frag_ld(lB + (wn * 64 + tn * 16 + (lane & 15)) * LDSS, lane);
#pragma unroll
    for (int tm = 0; tm < 2; ++tm)
#pragma unroll
      for (int tn = 0; tn < 4; ++tn)
        acc[tm][tn] = wmma_bf16(a[tm], b[tn], acc[tm][tn]);
  }

  const int rhi = (lane & 16) ? 8 : 0;
#pragma unroll
  for (int tm = 0; tm < 2; ++tm) {
#pragma unroll
    for (int tn = 0; tn < 4; ++tn) {
      int col = nbase + wn * 64 + tn * 16 + (lane & 15);
#pragma unroll
      for (int j = 0; j < 8; ++j) {
        int rloc = wm * 32 + tm * 16 + j + rhi;
        int s = srow[rloc];
        if (s >= 0) {
          ybuf[(size_t)s * DMODEL + col] = sgate[rloc] * acc[tm][tn][j];
        }
      }
    }
  }
}

// =====================================================================
// Phase 4: out[t] = Y[2t] + Y[2t+1]  (deterministic fixed-order combine)
// =====================================================================
__global__ __launch_bounds__(256) void moe_combine(
    const float* __restrict__ ybuf, float* __restrict__ out) {
  int i = blockIdx.x * 256 + threadIdx.x;       // < NTOK*DMODEL
  int t = i >> 10;                              // DMODEL == 1024
  int d = i & 1023;
  out[i] = ybuf[(size_t)(2 * t) * DMODEL + d] + ybuf[(size_t)(2 * t + 1) * DMODEL + d];
}

// =====================================================================
extern "C" void kernel_launch(void* const* d_in, const int* in_sizes, int n_in,
                              void* d_out, int out_size, void* d_ws, size_t ws_size,
                              hipStream_t stream) {
  (void)in_sizes; (void)n_in; (void)out_size; (void)ws_size;
  const float* x  = (const float*)d_in[0];   // [1,2048,1024]
  const float* wr = (const float*)d_in[1];   // [8,1024]
  const float* w1 = (const float*)d_in[2];   // [1024, 8*2048]
  const float* w2 = (const float*)d_in[3];   // [8*2048, 1024]
  float* out = (float*)d_out;

  char* ws = (char*)d_ws;
  size_t off = 0;
  int*    sel_slot = (int*)(ws + off);   off += (size_t)NEXP * NTOK * sizeof(int);     // 64 KB
  float*  sel_gate = (float*)(ws + off); off += (size_t)NEXP * NTOK * sizeof(float);   // 64 KB
  int*    cnt      = (int*)(ws + off);   off += 256;                                    // padded
  __bf16* hbuf     = (__bf16*)(ws + off); off += (size_t)(2 * NTOK) * DFFN * 2;         // 16 MB
  float*  ybuf     = (float*)(ws + off); off += (size_t)(2 * NTOK) * DMODEL * 4;        // 16 MB

  moe_router<<<dim3(1), dim3(256), 0, stream>>>(x, wr, sel_slot, sel_gate, cnt);

  dim3 g1(NTOK / MT, NEXP, DFFN / NT);   // (16, 8, 16)
  moe_gemm1<<<g1, dim3(256), 0, stream>>>(x, w1, sel_slot, cnt, hbuf);

  dim3 g2(NTOK / MT, NEXP, DMODEL / NT); // (16, 8, 8)
  moe_gemm2<<<g2, dim3(256), 0, stream>>>(hbuf, w2, sel_slot, sel_gate, cnt, ybuf);

  moe_combine<<<dim3((NTOK * DMODEL) / 256), dim3(256), 0, stream>>>(ybuf, out);
}
